// Dy_DownSample_30880814858242
// MI455X (gfx1250) — matrix operand order repaired and checked by reading, not compiled, not run on hardware
//
#include <hip/hip_runtime.h>
#include <hip/hip_bf16.h>
#include <math.h>

typedef __attribute__((ext_vector_type(16))) _Float16 v16h;
typedef __attribute__((ext_vector_type(8)))  float    v8f;

// Problem constants (fixed by the reference setup)
constexpr int  B_  = 16;
constexpr int  C_  = 64;
constexpr int  H_  = 256;
constexpr int  W_  = 256;
constexpr int  G_  = 4;            // GROUPS
constexpr int  h_  = H_ / 2;       // 128
constexpr int  w_  = W_ / 2;       // 128
constexpr long HW_ = (long)H_ * W_;

// ---------------------------------------------------------------------------
// Kernel 1: fused 1x1 convs (off + sc) via WMMA, sigmoid scaling, and
// pixel-unshuffle scatter into offg.
//
// GEMM view: A = [16 pixels x 64 ch] (f16), Bmat = [64 ch x 16 cols]
// (cols 0..3 = off0, off1, sc0, sc1; cols 4..15 zero), accumulated f32 via
// two v_wmma_f32_16x16x32_f16 steps (K = 2 x 32).
// One wave handles 16 consecutive pixels of one hi-res row.
// Weights are staged once per block into LDS so the B operand is built with
// branch-free ds_load + v_cndmask (no per-element EXEC toggling).
// ---------------------------------------------------------------------------
__global__ void __launch_bounds__(256)
conv_offset_wmma(const float* __restrict__ x,
                 const float* __restrict__ w_off,
                 const float* __restrict__ b_off,
                 const float* __restrict__ w_sc,
                 const float* __restrict__ b_sc,
                 float* __restrict__ offg) {
    __shared__ float tileC[8][16][4];   // [wave][pixel row M][col n]
    __shared__ float wlds[4][C_];       // rows: off0, off1, sc0, sc1

    // Stage all 4*64 weight floats with exactly one coalesced load per thread.
    {
        const int t = threadIdx.x;
        const int r = t >> 6;           // 0..3
        const int c = t & 63;           // 0..63
        const float* src = (r < 2) ? (w_off + r * C_) : (w_sc + (r - 2) * C_);
        wlds[r][c] = src[c];
    }
    __syncthreads();

    const int lane = threadIdx.x & 31;
    const int wave = threadIdx.x >> 5;
    const int row  = lane & 15;         // pixel row M this lane feeds
    const int hlf  = lane >> 4;         // 0: K {0-7,16-23}; 1: K {8-15,24-31}

    const long tile = (long)blockIdx.x * 8 + wave;   // 16 pixels per tile
    const long p0   = tile * 16;                     // flat pixel base
    const int  b    = (int)(p0 / HW_);
    const int  rem  = (int)(p0 - (long)b * HW_);
    const int  hy   = rem / W_;
    const int  hx0  = rem - hy * W_;                 // multiple of 16
    const int  hx   = hx0 + row;

    // Per-lane pixel base: x[b, 0, hy, hx]
    const float* xpix = x + (long)b * C_ * HW_ + (long)hy * W_ + hx;

    // B-matrix operands (loop-invariant): col = lane&15, K = e(0-7), 16+(e-8); lanes>=16: +8
    const int  ncol  = lane & 15;
    const bool live  = (ncol < 4);
    const int  nsel  = ncol & 3;
    const int  kb    = hlf * 16;
    v16h bmat[2];
    #pragma unroll
    for (int s = 0; s < 2; ++s) {
        #pragma unroll
        for (int e = 0; e < 16; ++e) {
            float wv = wlds[nsel][s * 32 + ((e < 8) ? e : 8 + e) + hlf * 8];   // unconditional ds_load
            bmat[s][e] = (_Float16)(live ? wv : 0.0f); // v_cndmask, no branch
        }
    }

    v8f acc = {};
    #pragma unroll
    for (int s = 0; s < 2; ++s) {       // K steps: channels [0,32) and [32,64)
        v16h a;
        #pragma unroll
        for (int e = 0; e < 16; ++e) {
            // A layout: lanes<16: K = e(0-7), 16+(e-8); lanes>=16: +8
            int k = s * 32 + ((e < 8) ? e : 8 + e) + hlf * 8;
            a[e] = (_Float16)xpix[(long)k * HW_];    // coalesced half-wave reads
        }
        acc = __builtin_amdgcn_wmma_f32_16x16x32_f16(
                  /*neg_a=*/false, a, /*neg_b=*/false, bmat[s],
                  /*c_mod=*/(short)0, acc, /*reuse_a=*/false, /*reuse_b=*/false);
    }

    // C layout: acc[v] -> (M = hlf*8 + v, N = lane&15)
    if (live) {
        #pragma unroll
        for (int v = 0; v < 8; ++v)
            tileC[wave][hlf * 8 + v][ncol] = acc[v];
    }
    __syncthreads();

    // Finalize: lane<16 -> c_orig 0 of pixel 'row'; lane>=16 -> c_orig 1.
    // off = sigmoid(conv_sc + b_sc) * 0.5 * (conv_off + b_off)
    const float offc = tileC[wave][row][hlf]     + b_off[hlf];
    const float scc  = tileC[wave][row][2 + hlf] + b_sc[hlf];
    const float sig  = 1.0f / (1.0f + expf(-scc));
    const float val  = sig * 0.5f * offc;

    // Pixel-unshuffle + rearrange:
    // ch = c_orig*4 + (hy&1)*2 + (hx&1); g = ch>>1 = c_orig*2 + (hy&1); two = hx&1
    // offg[(b*G+g), hh, ww, two]; inner index (ww*2+two) == hx -> contiguous stores
    const int g  = hlf * 2 + (hy & 1);
    const int hh = hy >> 1;
    offg[((long)(b * G_ + g) * h_ + hh) * (w_ * 2) + hx] = val;
}

// ---------------------------------------------------------------------------
// Kernel 2: bilinear grid-sample (border clamp). One block = one (b,g) plus
// an 8(hh) x 32(ww) tile; offsets + tap weights computed once and reused
// across the group's 16 channels. Output stores fully coalesced along ww.
// ---------------------------------------------------------------------------
__global__ void __launch_bounds__(256)
grid_sample_kernel(const float* __restrict__ x,
                   const float* __restrict__ offg,
                   float* __restrict__ out) {
    const int ww = blockIdx.x * 32 + threadIdx.x;
    const int hh = blockIdx.y * 8  + threadIdx.y;
    const int bg = blockIdx.z;
    const int b  = bg >> 2;
    const int g  = bg & 3;

    const float2 off2 =
        *(const float2*)(offg + (((long)bg * h_ + hh) * w_ + ww) * 2);

    // Reference math (note: grid x-coord uses h-position, y-coord uses w-pos)
    const float hpos = hh + 0.5f, wpos = ww + 0.5f;
    const float gx = 2.0f * (hpos + off2.x) / (float)w_ - 1.0f;
    const float gy = 2.0f * (wpos + off2.y) / (float)h_ - 1.0f;
    float ix = ((gx + 1.0f) * (float)W_ - 1.0f) * 0.5f;
    float iy = ((gy + 1.0f) * (float)H_ - 1.0f) * 0.5f;
    ix = fminf(fmaxf(ix, 0.0f), (float)(W_ - 1));
    iy = fminf(fmaxf(iy, 0.0f), (float)(H_ - 1));

    const float x0f = floorf(ix), y0f = floorf(iy);
    const float wx = ix - x0f, wy = iy - y0f;
    const int x0 = (int)x0f, y0 = (int)y0f;
    const int x1 = min(x0 + 1, W_ - 1), y1 = min(y0 + 1, H_ - 1);

    const float c00 = (1.0f - wx) * (1.0f - wy);
    const float c01 = wx * (1.0f - wy);
    const float c10 = (1.0f - wx) * wy;
    const float c11 = wx * wy;

    const long o00 = (long)y0 * W_ + x0;
    const long o01 = (long)y0 * W_ + x1;
    const long o10 = (long)y1 * W_ + x0;
    const long o11 = (long)y1 * W_ + x1;

    const float* xb = x + (long)b * C_ * HW_;
    #pragma unroll
    for (int c = 0; c < C_ / G_; ++c) {
        const int ci = c * G_ + g;                   // xg channel == out channel
        const float* pl = xb + (long)ci * HW_;
        const float v = pl[o00] * c00 + pl[o01] * c01 +
                        pl[o10] * c10 + pl[o11] * c11;
        out[(((long)b * C_ + ci) * h_ + hh) * w_ + ww] = v;
    }
}

extern "C" void kernel_launch(void* const* d_in, const int* in_sizes, int n_in,
                              void* d_out, int out_size, void* d_ws, size_t ws_size,
                              hipStream_t stream) {
    const float* x     = (const float*)d_in[0];
    const float* w_off = (const float*)d_in[1];
    const float* b_off = (const float*)d_in[2];
    const float* w_sc  = (const float*)d_in[3];
    const float* b_sc  = (const float*)d_in[4];

    float* out  = (float*)d_out;                       // (B, C, h, w)
    float* offg = out + (long)B_ * C_ * h_ * w_;       // (B*G, h, w, 2) tail

    // Kernel 1: B*H*W pixels, 16 per wave, 8 waves per block
    const int n_tiles  = (int)((long)B_ * H_ * W_ / 16);   // 65536
    const int n_blocks = n_tiles / 8;                      // 8192
    conv_offset_wmma<<<n_blocks, 256, 0, stream>>>(x, w_off, b_off, w_sc, b_sc, offg);

    // Kernel 2: grid over (ww/32, hh/8, B*G)
    dim3 grid2(w_ / 32, h_ / 8, B_ * G_);
    dim3 blk2(32, 8, 1);
    grid_sample_kernel<<<grid2, blk2, 0, stream>>>(x, offg, out);
}
